// DGCNNv2Seg_60687887892807
// MI455X (gfx1250) — matrix operand order repaired and checked by reading, not compiled, not run on hardware
//
#include <hip/hip_runtime.h>

// DGCNN part-seg forward for MI455X (gfx1250, wave32, WMMA).
// GEMMs use v_wmma_f32_16x16x32_f16 with operands pre-tiled into a
// per-lane fragment layout (k = 16*half + e, same for A and B): each fragment is one contiguous
// 32B-per-lane load (2x global_load_b128). 1x4 column blocking per wave and
// copy-free ping-pong double buffering over k-chunks (unroll-by-2) so loads
// overlap the WMMA pipe without register rotation moves.
// EdgeConv BN+LeakyReLU+max-over-K fused via float atomicMax.
// Workspace requirement: ~120 MB.

#define NPTS  4096
#define KNN   20
#define EPSB  1e-5f
#define SLOPE 0.2f
#define WAVES 4
#define TPB   (WAVES * 32)
#define CTILE 4   // column tiles (of 16) per wave

typedef __attribute__((ext_vector_type(16))) _Float16 v16h;
typedef __attribute__((ext_vector_type(8)))  float    v8f;

// ---------------------------------------------------------------------------
// Fragment-layout index helpers (lane half h, element e holds k = 16h + e; wave32).
__device__ __forceinline__ size_t b_tiled_index(int i, size_t j, size_t J) {
  const int kc   = i >> 5;
  const int rem  = i & 31;
  const int half = rem >> 4;
  const int e    = rem & 15;
  const size_t jt = j >> 4;
  const int lane = half * 16 + (int)(j & 15);
  const size_t numJT = J >> 4;
  return (((size_t)kc * numJT + jt) * 32 + lane) * 16 + e;
}

__device__ __forceinline__ size_t a_tiled_index(int o, int i, int Ipad) {
  const int ot = o >> 4, m = o & 15;
  const int kc = i >> 5, kb = i & 31;
  const int half = kb >> 4;
  const int e    = kb & 15;
  const int lane = half * 16 + m;
  const int numKC = Ipad >> 5;
  return (((size_t)ot * numKC + kc) * 32 + lane) * 16 + e;
}

__device__ __forceinline__ void load_frag_set(const v16h* __restrict__ A,
                                              const v16h* __restrict__ B,
                                              int rt, int numKC, size_t numJT, size_t jt0,
                                              int lane, int kc, v16h& a, v16h b[CTILE]) {
  a = A[((size_t)rt * numKC + kc) * 32 + lane];
  const v16h* bp = B + (((size_t)kc * numJT + jt0) * 32) + lane;
#pragma unroll
  for (int ct = 0; ct < CTILE; ++ct) b[ct] = bp[(size_t)ct * 32];
}

__device__ __forceinline__ void wmma_set(const v16h& a, const v16h b[CTILE], v8f acc[CTILE]) {
#pragma unroll
  for (int ct = 0; ct < CTILE; ++ct)
    acc[ct] = __builtin_amdgcn_wmma_f32_16x16x32_f16(false, a, false, b[ct],
                                                     (short)0, acc[ct], false, false);
}

// Ping-pong pipelined fragment GEMM core: 16x64 strip per wave, no reg copies.
__device__ __forceinline__ void gemm_core(const v16h* __restrict__ A,
                                          const v16h* __restrict__ B,
                                          int rt, size_t jt0, size_t numJT, int numKC,
                                          int lane, v8f acc[CTILE]) {
  v16h a0, a1, b0[CTILE], b1[CTILE];
  load_frag_set(A, B, rt, numKC, numJT, jt0, lane, 0, a0, b0);
  int kc = 0;
  for (; kc + 2 <= numKC; kc += 2) {
    load_frag_set(A, B, rt, numKC, numJT, jt0, lane, kc + 1, a1, b1);
    if (kc + 2 < numKC)
      __builtin_prefetch((const void*)(B + (((size_t)(kc + 2) * numJT + jt0) * 32) + lane),
                         0, 3);  // WGP-scope global_prefetch_b8
    wmma_set(a0, b0, acc);      // consume set0 while set1 loads are in flight
    if (kc + 2 < numKC)
      load_frag_set(A, B, rt, numKC, numJT, jt0, lane, kc + 2, a0, b0);
    wmma_set(a1, b1, acc);      // consume set1 while set0 loads are in flight
  }
  if (kc < numKC)
    wmma_set(a0, b0, acc);      // odd tail (numKC == 1 for layer 1)
}

// ---------------------------------------------------------------------------
// EdgeConv GEMM: out[o,n] = max_k lrelu(bn(sum_i W[o,i]*F[i, n*K+k])).
__global__ void __launch_bounds__(TPB)
gemm_edge_wmma(const v16h* __restrict__ A, const v16h* __restrict__ B,
               float* __restrict__ out, int Ipad, int rowTiles,
               const float* __restrict__ bng, const float* __restrict__ bnb,
               const float* __restrict__ bnm, const float* __restrict__ bnv) {
  const size_t J = (size_t)NPTS * KNN;
  const size_t numJT = J >> 4;
  const int numKC = Ipad >> 5;
  const int lane = threadIdx.x & 31;
  const int rt = blockIdx.y * WAVES + (threadIdx.x >> 5);
  if (rt >= rowTiles) return;  // uniform per wave
  const size_t jt0 = (size_t)blockIdx.x * CTILE;

  v8f acc[CTILE] = {};
  gemm_core(A, B, rt, jt0, numJT, numKC, lane, acc);

  const int half = lane >> 4, n = lane & 15;
#pragma unroll
  for (int ct = 0; ct < CTILE; ++ct) {
    const size_t j = (jt0 + ct) * 16 + (size_t)n;
    const int npt = (int)(j / KNN);
#pragma unroll
    for (int r = 0; r < 8; ++r) {
      const int o = rt * 16 + r + 8 * half;
      const float sc = bng[o] * rsqrtf(bnv[o] + EPSB);
      float y = (acc[ct][r] - bnm[o]) * sc + bnb[o];
      y = (y >= 0.0f) ? y : SLOPE * y;
      atomicMax(&out[(size_t)o * NPTS + npt], y);  // global_atomic_max_num_f32
    }
  }
}

// Pointwise conv GEMM: out[o,n] = lrelu(bn(sum_i W[o,i]*F[i,n])), J = NPTS.
__global__ void __launch_bounds__(TPB)
gemm_conv_wmma(const v16h* __restrict__ A, const v16h* __restrict__ B,
               float* __restrict__ out, int Ipad, int rowTiles,
               const float* __restrict__ bng, const float* __restrict__ bnb,
               const float* __restrict__ bnm, const float* __restrict__ bnv) {
  const size_t J = (size_t)NPTS;
  const size_t numJT = J >> 4;
  const int numKC = Ipad >> 5;
  const int lane = threadIdx.x & 31;
  const int rt = blockIdx.y * WAVES + (threadIdx.x >> 5);
  if (rt >= rowTiles) return;
  const size_t jt0 = (size_t)blockIdx.x * CTILE;

  v8f acc[CTILE] = {};
  gemm_core(A, B, rt, jt0, numJT, numKC, lane, acc);

  const int half = lane >> 4, n = lane & 15;
#pragma unroll
  for (int ct = 0; ct < CTILE; ++ct) {
    const size_t j = (jt0 + ct) * 16 + (size_t)n;
#pragma unroll
    for (int r = 0; r < 8; ++r) {
      const int o = rt * 16 + r + 8 * half;
      const float sc = bng[o] * rsqrtf(bnv[o] + EPSB);
      float y = (acc[ct][r] - bnm[o]) * sc + bnb[o];
      y = (y >= 0.0f) ? y : SLOPE * y;
      out[(size_t)o * NPTS + j] = y;
    }
  }
}

// ---------------------------------------------------------------------------
// Support kernels.
__global__ void transpose_points(const float* __restrict__ pts, float* __restrict__ xT) {
  const int n = blockIdx.x * 256 + threadIdx.x;
  const int c = blockIdx.y;
  xT[(size_t)c * NPTS + n] = pts[(size_t)n * 3 + c];
}

__global__ void sumsq_kernel(const float* __restrict__ x, int C, float* __restrict__ xx) {
  const int n = blockIdx.x * 256 + threadIdx.x;
  float s = 0.f;
  for (int c = 0; c < C; ++c) {
    const float v = x[(size_t)c * NPTS + n];
    s += v * v;
  }
  xx[n] = s;
}

// Brute-force top-K (descending neg-distance, stable on ties -> matches top_k).
__global__ void knn_kernel(const float* __restrict__ x, int C,
                           const float* __restrict__ xx, int* __restrict__ idx) {
  const int n = blockIdx.x * 256 + threadIdx.x;
  float bd[KNN];
  int   bi[KNN];
#pragma unroll
  for (int k = 0; k < KNN; ++k) { bd[k] = -INFINITY; bi[k] = 0; }
  const float xxn = xx[n];
  for (int m = 0; m < NPTS; ++m) {
    float inner = 0.f;
    for (int c = 0; c < C; ++c)
      inner += x[(size_t)c * NPTS + n] * x[(size_t)c * NPTS + m];
    const float nd = 2.f * inner - xxn - xx[m];
    if (nd > bd[KNN - 1]) {
      int p = KNN - 1;
      while (p > 0 && bd[p - 1] < nd) {
        bd[p] = bd[p - 1]; bi[p] = bi[p - 1]; --p;
      }
      bd[p] = nd; bi[p] = m;
    }
  }
  for (int k = 0; k < KNN; ++k) idx[(size_t)n * KNN + k] = bi[k];
}

// Edge features (fragment layout): rows [0,C)=x[:,nb]-x[:,n]; [C,2C)=x[:,n].
__global__ void build_feature_f16(const float* __restrict__ x, const int* __restrict__ idx,
                                  int C, _Float16* __restrict__ F) {
  const size_t J = (size_t)NPTS * KNN;
  const size_t j = (size_t)blockIdx.x * 256 + threadIdx.x;
  const int i = blockIdx.y;
  if (j >= J) return;
  const int n = (int)(j / KNN);
  float val = 0.f;
  if (i < C) {
    const int nb = idx[j];
    val = x[(size_t)i * NPTS + nb] - x[(size_t)i * NPTS + n];
  } else if (i < 2 * C) {
    val = x[(size_t)(i - C) * NPTS + n];
  }
  F[b_tiled_index(i, j, J)] = (_Float16)val;
}

__global__ void pad_weight_f16(const float* __restrict__ w, int O, int I, int Ipad,
                               _Float16* __restrict__ wp) {
  const int t = blockIdx.x * 256 + threadIdx.x;
  if (t >= O * Ipad) return;
  const int o = t / Ipad, i = t % Ipad;
  wp[a_tiled_index(o, i, Ipad)] = (_Float16)((i < I) ? w[(size_t)o * I + i] : 0.f);
}

__global__ void fill_kernel(float* __restrict__ p, size_t n, float v) {
  const size_t t = (size_t)blockIdx.x * 256 + threadIdx.x;
  if (t < n) p[t] = v;
}

// Row-major f32 (rows x NPTS) -> fragment-layout f16 panel rows [rowOff, ...).
__global__ void pack_tiled_f16(const float* __restrict__ s, _Float16* __restrict__ panel,
                               int rowOff) {
  const size_t j = (size_t)blockIdx.x * 256 + threadIdx.x;
  const int r = blockIdx.y;
  panel[b_tiled_index(rowOff + r, j, (size_t)NPTS)] =
      (_Float16)s[(size_t)r * NPTS + j];
}

__global__ void colmax_kernel(const float* __restrict__ x, float* __restrict__ g) {
  __shared__ float red[256];
  const int c = blockIdx.x, tid = threadIdx.x;
  float mx = -INFINITY;
  for (int n = tid; n < NPTS; n += 256) mx = fmaxf(mx, x[(size_t)c * NPTS + n]);
  red[tid] = mx;
  __syncthreads();
  for (int s = 128; s > 0; s >>= 1) {
    if (tid < s) red[tid] = fmaxf(red[tid], red[tid + s]);
    __syncthreads();
  }
  if (tid == 0) g[c] = red[0];
}

// Broadcast global-max vector into fragment-layout rows [512,1024) of panel.
__global__ void bcast_g16(const float* __restrict__ g, _Float16* __restrict__ panel) {
  const int n = blockIdx.x * 256 + threadIdx.x;
  const int r = blockIdx.y;
  panel[b_tiled_index(512 + r, (size_t)n, (size_t)NPTS)] = (_Float16)g[r];
}

__global__ void final_logits(const float* __restrict__ h, const float* __restrict__ w,
                             const float* __restrict__ bias, float* __restrict__ out) {
  const int n = blockIdx.x * 256 + threadIdx.x;
  const int o = blockIdx.y;
  float s = bias[o];
  for (int i = 0; i < 128; ++i)
    s += w[(size_t)o * 128 + i] * h[(size_t)i * NPTS + n];
  out[(size_t)o * NPTS + n] = s;
}

// ---------------------------------------------------------------------------
extern "C" void kernel_launch(void* const* d_in, const int* in_sizes, int n_in,
                              void* d_out, int out_size, void* d_ws, size_t ws_size,
                              hipStream_t stream) {
  (void)in_sizes; (void)n_in; (void)out_size; (void)ws_size;
  const float* points = (const float*)d_in[0];
  const float* Wsrc[7] = { (const float*)d_in[1],  (const float*)d_in[6],
                           (const float*)d_in[11], (const float*)d_in[16],
                           (const float*)d_in[21], (const float*)d_in[26],
                           (const float*)d_in[31] };
  const int WO[7]  = { 64, 64, 128, 256, 512, 256, 128 };
  const int WI[7]  = { 6, 128, 128, 256, 512, 1024, 256 };
  const int WIp[7] = { 32, 128, 128, 256, 512, 1024, 256 };
  const int BNbase[7] = { 2, 7, 12, 17, 22, 27, 32 };
  const float* ws3 = (const float*)d_in[36];
  const float* bs3 = (const float*)d_in[37];

  char* wsp = (char*)d_ws;
  auto take = [&](size_t bytes) -> char* {
    char* r = wsp;
    wsp += (bytes + 255) & ~(size_t)255;
    return r;
  };
  _Float16* Wp[7];
  for (int i = 0; i < 7; ++i)
    Wp[i] = (_Float16*)take((size_t)WO[i] * WIp[i] * sizeof(_Float16));
  float*    xT   = (float*)take((size_t)3 * NPTS * 4);
  float*    xx   = (float*)take((size_t)NPTS * 4);
  int*      nidx = (int*)take((size_t)NPTS * KNN * 4);
  float*    xcat = (float*)take((size_t)512 * NPTS * 4);
  _Float16* feat = (_Float16*)take((size_t)512 * NPTS * KNN * 2);  // 84 MB
  _Float16* c16  = (_Float16*)take((size_t)1024 * NPTS * 2);
  float*    x5   = (float*)take((size_t)512 * NPTS * 4);
  float*    gmax = (float*)take((size_t)512 * 4);
  float*    h1   = (float*)take((size_t)256 * NPTS * 4);
  _Float16* h116 = (_Float16*)take((size_t)256 * NPTS * 2);
  float*    h2   = (float*)take((size_t)128 * NPTS * 4);

  for (int i = 0; i < 7; ++i) {
    const int cnt = WO[i] * WIp[i];
    pad_weight_f16<<<(cnt + 255) / 256, 256, 0, stream>>>(Wsrc[i], WO[i], WI[i], WIp[i], Wp[i]);
  }

  auto bn = [&](int layer, int slot) -> const float* {
    return (const float*)d_in[BNbase[layer] + slot];
  };

  auto edge = [&](const float* xin, int C, int layer, float* outRegion) {
    const int Ipad = WIp[layer], O = WO[layer], rowTiles = O / 16;
    sumsq_kernel<<<NPTS / 256, 256, 0, stream>>>(xin, C, xx);
    knn_kernel<<<NPTS / 256, 256, 0, stream>>>(xin, C, xx, nidx);
    build_feature_f16<<<dim3((NPTS * KNN + 255) / 256, Ipad), 256, 0, stream>>>(xin, nidx, C, feat);
    fill_kernel<<<(unsigned)(((size_t)O * NPTS + 255) / 256), 256, 0, stream>>>(
        outRegion, (size_t)O * NPTS, -INFINITY);
    gemm_edge_wmma<<<dim3((NPTS * KNN) / (16 * CTILE), (rowTiles + WAVES - 1) / WAVES),
                     TPB, 0, stream>>>(
        (const v16h*)Wp[layer], (const v16h*)feat, outRegion, Ipad, rowTiles,
        bn(layer, 0), bn(layer, 1), bn(layer, 2), bn(layer, 3));
  };

  auto conv = [&](const _Float16* fin, int layer, float* outBuf) {
    const int Ipad = WIp[layer], rowTiles = WO[layer] / 16;
    gemm_conv_wmma<<<dim3(NPTS / (16 * CTILE), (rowTiles + WAVES - 1) / WAVES),
                     TPB, 0, stream>>>(
        (const v16h*)Wp[layer], (const v16h*)fin, outBuf, Ipad, rowTiles,
        bn(layer, 0), bn(layer, 1), bn(layer, 2), bn(layer, 3));
  };

  for (int b = 0; b < 2; ++b) {
    const float* pts = points + (size_t)b * NPTS * 3;
    float* outb = (float*)d_out + (size_t)b * 2 * NPTS;

    transpose_points<<<dim3(NPTS / 256, 3), 256, 0, stream>>>(pts, xT);

    // EdgeConvs write directly into their concat rows of xcat.
    edge(xT, 3, 0, xcat);                                               // rows 0..63
    edge(xcat, 64, 1, xcat + (size_t)64 * NPTS);                        // rows 64..127
    edge(xcat + (size_t)64 * NPTS, 64, 2, xcat + (size_t)128 * NPTS);   // rows 128..255
    edge(xcat + (size_t)128 * NPTS, 128, 3, xcat + (size_t)256 * NPTS); // rows 256..511

    // x5 = lrelu(bn(w5 @ xcat)); global max over N.
    pack_tiled_f16<<<dim3(NPTS / 256, 512), 256, 0, stream>>>(xcat, c16, 0);
    conv(c16, 4, x5);
    colmax_kernel<<<512, 256, 0, stream>>>(x5, gmax);

    // feat2 = [xcat ; broadcast(gmax)] in fragment-layout f16.
    bcast_g16<<<dim3(NPTS / 256, 512), 256, 0, stream>>>(gmax, c16);
    conv(c16, 5, h1);  // ws1: 256 x 1024
    pack_tiled_f16<<<dim3(NPTS / 256, 256), 256, 0, stream>>>(h1, h116, 0);
    conv(h116, 6, h2); // ws2: 128 x 256

    final_logits<<<dim3(NPTS / 256, 2), 256, 0, stream>>>(h2, ws3, bs3, outb);
  }
}
